// LocalMPNNLayer_13950053777620
// MI455X (gfx1250) — matrix operand order repaired and checked by reading, not compiled, not run on hardware
//
#include <hip/hip_runtime.h>
#include <hip/hip_bf16.h>

typedef __attribute__((ext_vector_type(16))) _Float16 v16h;
typedef __attribute__((ext_vector_type(8)))  float    v8f;

union V16H { v16h v; _Float16 e[16]; };
union V8F  { v8f  v; float    e[8];  };

#define D_      128
#define NRBF    20
#define KNN     16
#define CUT     5.0f
#define NNODES  4096
#define NBATCH  8

// A-operand k order used by this code (16x32 MxK tile; B staged the same):
// lanes 0-15: M=lane, halves 0..15 -> K 0..15
// lanes 16-31: M=lane-16, halves 0..15 -> K 16..31
__device__ __forceinline__ int a_klocal(int lane, int q) {
    return q + ((lane >> 4) << 4);
}

__device__ __forceinline__ float silu_f(float x) {
    return x / (1.0f + __expf(-x));
}

// ---------------------------------------------------------------------------
// Kernel 1: kNN graph build. One block (128 threads) per (b,i) row.
// ---------------------------------------------------------------------------
__global__ void knn_kernel(const float* __restrict__ pos,
                           int* __restrict__ eidx,
                           float* __restrict__ edist) {
    __shared__ float sd[NNODES];
    __shared__ float rmin[128];
    __shared__ int   rarg[128];
    const int tid = threadIdx.x;
    const int row = blockIdx.x;              // b*N + i
    const int b   = row / NNODES;
    const int i   = row - b * NNODES;
    const float xi = pos[(size_t)row * 3 + 0];
    const float yi = pos[(size_t)row * 3 + 1];
    const float zi = pos[(size_t)row * 3 + 2];
    const float* pb = pos + (size_t)b * NNODES * 3;
    for (int j = tid; j < NNODES; j += 128) {
        float dx = xi - pb[j * 3 + 0];
        float dy = yi - pb[j * 3 + 1];
        float dz = zi - pb[j * 3 + 2];
        float d  = sqrtf(dx * dx + dy * dy + dz * dz + 1e-8f);
        sd[j] = (j == i) ? 1e30f : d;
    }
    __syncthreads();
    for (int r = 0; r < KNN; ++r) {
        float best = 1e30f; int bj = 0;
        for (int j = tid; j < NNODES; j += 128) {
            float d = sd[j];
            if (d < best) { best = d; bj = j; }
        }
        rmin[tid] = best; rarg[tid] = bj;
        __syncthreads();
        if (tid == 0) {
            float bb = 1e30f; int bi2 = 0;
            for (int t = 0; t < 128; ++t)
                if (rmin[t] < bb) { bb = rmin[t]; bi2 = rarg[t]; }
            eidx[(size_t)row * KNN + r]  = b * NNODES + bi2;   // flat neighbor id
            edist[(size_t)row * KNN + r] = bb;
            sd[bi2] = 1e30f;
        }
        __syncthreads();
    }
}

// Scatter a C-fragment value (M=r+Mr_hi, N) into an A-fragment-major f16 LDS
// buffer laid out as frag[(kt*32 + lane)*16 + q], so the next layer's
// A-operand is one aligned v16h load.
__device__ __forceinline__ int afrag_addr(int M, int N) {
    const int kt    = N >> 5;
    const int klmod = N & 31;
    const int lp    = M + ((klmod & 16) ? 16 : 0);
    const int q     = klmod & 15;
    return (kt * 32 + lp) * 16 + q;
}

// ---------------------------------------------------------------------------
// Kernel 2: edge MLP (WMMA) + mask + per-node aggregate.
// Weights pre-swizzled into fragment-major f16 LDS: one v16h load per B frag.
// ---------------------------------------------------------------------------
__global__ void edge_mlp_kernel(const float* __restrict__ h,
                                const int* __restrict__ eidx,
                                const float* __restrict__ edist,
                                const float* __restrict__ ew1,
                                const float* __restrict__ eb1,
                                const float* __restrict__ ew2,
                                const float* __restrict__ eb2,
                                float* __restrict__ agg,
                                int total_nodes) {
    extern __shared__ __align__(32) char smem[];
    _Float16* w1f = (_Float16*)smem;                // 9*8 tiles * 512 halves
    _Float16* w2f = w1f + 9 * 8 * 512;              // 4*8 tiles * 512 halves
    char* pw = (char*)(w2f + 4 * 8 * 512);

    const int tid    = threadIdx.x;
    const int lane   = tid & 31;
    const int warp   = tid >> 5;
    const int nwaves = blockDim.x >> 5;

    char*     base   = pw + warp * 4352;
    _Float16* m1f    = (_Float16*)base;             // 4*32*16 halves (frag-major)
    int*      myidx  = (int*)(base + 4096);         // [16]
    float*    mydist = (float*)(base + 4096 + 64);  // [16]
    float*    mymask = (float*)(base + 4096 + 128); // [16]

    // ---- stage weights, fragment-major: wf[(kt*8+t)*512 + lane_*16 + q] =
    //      W[kt*32 + q + 16*(lane_>>4)][t*16 + (lane_&15)]
    for (int d = tid; d < 9 * 8 * 512; d += blockDim.x) {
        const int tile  = d >> 9;
        const int lane_ = (d >> 4) & 31;
        const int q     = d & 15;
        const int k = (tile >> 3) * 32 + q + ((lane_ >> 4) << 4);
        const int n = (tile & 7) * 16 + (lane_ & 15);
        w1f[d] = (_Float16)((k < 276) ? ew1[k * 128 + n] : 0.0f);
    }
    for (int d = tid; d < 4 * 8 * 512; d += blockDim.x) {
        const int tile  = d >> 9;
        const int lane_ = (d >> 4) & 31;
        const int q     = d & 15;
        const int k = (tile >> 3) * 32 + q + ((lane_ >> 4) << 4);
        const int n = (tile & 7) * 16 + (lane_ & 15);
        w2f[d] = (_Float16)ew2[k * 128 + n];
    }
    __syncthreads();

    const int waveGlobal = blockIdx.x * nwaves + warp;
    const int waveTotal  = gridDim.x * nwaves;
    const int M      = lane & 15;
    const int Mr_hi  = (lane >> 4) << 3;            // 0 or 8

    for (int node = waveGlobal; node < total_nodes; node += waveTotal) {
        if (node + waveTotal < total_nodes)
            __builtin_prefetch(h + (size_t)(node + waveTotal) * D_, 0, 1);

        if (lane < KNN) {
            int   j = eidx[(size_t)node * KNN + lane];
            float d = edist[(size_t)node * KNN + lane];
            myidx[lane]  = j;
            mydist[lane] = d;
            mymask[lane] = (d < CUT) ? 1.0f : 0.0f;
        }

        // ---- layer 1: [16x288] @ [288x128] -> silu -> m1 (A-frag layout) ----
        V8F acc[8];
        #pragma unroll
        for (int t = 0; t < 8; ++t)
            #pragma unroll
            for (int r = 0; r < 8; ++r) acc[t].e[r] = 0.0f;

        #pragma unroll
        for (int kt = 0; kt < 9; ++kt) {
            V16H a;
            if (kt < 4) {                                 // h_i block
                const float* src = h + (size_t)node * D_ + kt * 32;
                #pragma unroll
                for (int q = 0; q < 16; ++q) a.e[q] = (_Float16)src[a_klocal(lane, q)];
            } else if (kt < 8) {                          // h_j block
                const int j = myidx[M];
                const float* src = h + (size_t)j * D_ + (kt - 4) * 32;
                #pragma unroll
                for (int q = 0; q < 16; ++q) a.e[q] = (_Float16)src[a_klocal(lane, q)];
            } else {                                      // RBF block (+ zero pad)
                const float d = mydist[M];
                #pragma unroll
                for (int q = 0; q < 16; ++q) {
                    int kl = a_klocal(lane, q);
                    float v = 0.0f;
                    if (kl < NRBF) {
                        float c = CUT * (float)kl / (float)(NRBF - 1);
                        float u = d - c;
                        v = __expf(-16.0f * u * u);       // widths = (CUT/NRBF)^-2
                    }
                    a.e[q] = (_Float16)v;
                }
            }
            #pragma unroll
            for (int t = 0; t < 8; ++t) {
                V16H bf;
                bf.v = *(const v16h*)(w1f + (size_t)(kt * 8 + t) * 512 + lane * 16);
                acc[t].v = __builtin_amdgcn_wmma_f32_16x16x32_f16(
                    false, a.v, false, bf.v, (short)0, acc[t].v, false, false);
            }
        }
        #pragma unroll
        for (int t = 0; t < 8; ++t) {
            const int   N  = t * 16 + (lane & 15);
            const float b1 = eb1[N];
            #pragma unroll
            for (int r = 0; r < 8; ++r)
                m1f[afrag_addr(r + Mr_hi, N)] = (_Float16)silu_f(acc[t].e[r] + b1);
        }

        // ---- layer 2: [16x128] @ [128x128] -> silu -> mask -> column sum ----
        V8F acc2[8];
        #pragma unroll
        for (int t = 0; t < 8; ++t)
            #pragma unroll
            for (int r = 0; r < 8; ++r) acc2[t].e[r] = 0.0f;

        #pragma unroll
        for (int kt = 0; kt < 4; ++kt) {
            V16H a;
            a.v = *(const v16h*)(m1f + (kt * 32 + lane) * 16);
            #pragma unroll
            for (int t = 0; t < 8; ++t) {
                V16H bf;
                bf.v = *(const v16h*)(w2f + (size_t)(kt * 8 + t) * 512 + lane * 16);
                acc2[t].v = __builtin_amdgcn_wmma_f32_16x16x32_f16(
                    false, a.v, false, bf.v, (short)0, acc2[t].v, false, false);
            }
        }
        #pragma unroll
        for (int t = 0; t < 8; ++t) {
            const int   N  = t * 16 + (lane & 15);
            const float b2 = eb2[N];
            float col = 0.0f;
            #pragma unroll
            for (int r = 0; r < 8; ++r)
                col += silu_f(acc2[t].e[r] + b2) * mymask[r + Mr_hi];
            col += __shfl_xor(col, 16, 32);               // combine M 0-7 with M 8-15
            if (lane < 16)
                agg[(size_t)node * D_ + t * 16 + lane] = col;
        }
    }
}

// ---------------------------------------------------------------------------
// Kernel 3: node MLP (WMMA) + residual + LayerNorm. One wave per 16-node tile.
// ---------------------------------------------------------------------------
__global__ void node_kernel(const float* __restrict__ h,
                            const float* __restrict__ agg,
                            const float* __restrict__ nw1,
                            const float* __restrict__ nb1,
                            const float* __restrict__ nw2,
                            const float* __restrict__ nb2,
                            const float* __restrict__ gamma,
                            const float* __restrict__ beta,
                            float* __restrict__ out,
                            int ntiles) {
    extern __shared__ __align__(32) char smem[];
    _Float16* w1f = (_Float16*)smem;                // 8*8 tiles * 512 halves
    _Float16* w2f = w1f + 8 * 8 * 512;              // 4*8 tiles * 512 halves
    char* pw = (char*)(w2f + 4 * 8 * 512);

    const int tid    = threadIdx.x;
    const int lane   = tid & 31;
    const int warp   = tid >> 5;
    const int nwaves = blockDim.x >> 5;

    char*     base = pw + warp * 12288;
    _Float16* u1f  = (_Float16*)base;               // 4*32*16 halves (frag-major)
    float*    xs   = (float*)(base + 4096);         // [16][128] f32

    for (int d = tid; d < 8 * 8 * 512; d += blockDim.x) {
        const int tile  = d >> 9;
        const int lane_ = (d >> 4) & 31;
        const int q     = d & 15;
        const int k = (tile >> 3) * 32 + q + ((lane_ >> 4) << 4);
        const int n = (tile & 7) * 16 + (lane_ & 15);
        w1f[d] = (_Float16)nw1[k * 128 + n];
    }
    for (int d = tid; d < 4 * 8 * 512; d += blockDim.x) {
        const int tile  = d >> 9;
        const int lane_ = (d >> 4) & 31;
        const int q     = d & 15;
        const int k = (tile >> 3) * 32 + q + ((lane_ >> 4) << 4);
        const int n = (tile & 7) * 16 + (lane_ & 15);
        w2f[d] = (_Float16)nw2[k * 128 + n];
    }
    __syncthreads();

    const int waveGlobal = blockIdx.x * nwaves + warp;
    const int waveTotal  = gridDim.x * nwaves;
    const int M     = lane & 15;
    const int Mr_hi = (lane >> 4) << 3;

    for (int tile = waveGlobal; tile < ntiles; tile += waveTotal) {
        const int ng = tile * 16 + M;               // row for A-fragment loads

        // ---- layer 1: [16x256] @ [256x128] -> +nb1 -> silu -> u1 ----
        V8F acc[8];
        #pragma unroll
        for (int t = 0; t < 8; ++t)
            #pragma unroll
            for (int r = 0; r < 8; ++r) acc[t].e[r] = 0.0f;

        #pragma unroll
        for (int kt = 0; kt < 8; ++kt) {
            V16H a;
            const float* src = (kt < 4) ? (h   + (size_t)ng * D_ + kt * 32)
                                        : (agg + (size_t)ng * D_ + (kt - 4) * 32);
            #pragma unroll
            for (int q = 0; q < 16; ++q) a.e[q] = (_Float16)src[a_klocal(lane, q)];
            #pragma unroll
            for (int t = 0; t < 8; ++t) {
                V16H bf;
                bf.v = *(const v16h*)(w1f + (size_t)(kt * 8 + t) * 512 + lane * 16);
                acc[t].v = __builtin_amdgcn_wmma_f32_16x16x32_f16(
                    false, a.v, false, bf.v, (short)0, acc[t].v, false, false);
            }
        }
        #pragma unroll
        for (int t = 0; t < 8; ++t) {
            const int   N  = t * 16 + (lane & 15);
            const float b1 = nb1[N];
            #pragma unroll
            for (int r = 0; r < 8; ++r)
                u1f[afrag_addr(r + Mr_hi, N)] = (_Float16)silu_f(acc[t].e[r] + b1);
        }

        // ---- layer 2: [16x128] @ [128x128] -> +nb2 + residual -> xs ----
        V8F acc2[8];
        #pragma unroll
        for (int t = 0; t < 8; ++t)
            #pragma unroll
            for (int r = 0; r < 8; ++r) acc2[t].e[r] = 0.0f;

        #pragma unroll
        for (int kt = 0; kt < 4; ++kt) {
            V16H a;
            a.v = *(const v16h*)(u1f + (kt * 32 + lane) * 16);
            #pragma unroll
            for (int t = 0; t < 8; ++t) {
                V16H bf;
                bf.v = *(const v16h*)(w2f + (size_t)(kt * 8 + t) * 512 + lane * 16);
                acc2[t].v = __builtin_amdgcn_wmma_f32_16x16x32_f16(
                    false, a.v, false, bf.v, (short)0, acc2[t].v, false, false);
            }
        }
        #pragma unroll
        for (int t = 0; t < 8; ++t) {
            const int   N  = t * 16 + (lane & 15);
            const float b2 = nb2[N];
            #pragma unroll
            for (int r = 0; r < 8; ++r) {
                const int Mr  = r + Mr_hi;
                const int ngr = tile * 16 + Mr;
                xs[Mr * D_ + N] = acc2[t].e[r] + b2 + h[(size_t)ngr * D_ + N];
            }
        }

        // ---- LayerNorm: 2 lanes per row, 64 columns each ----
        const int row  = lane >> 1;
        const int hsel = lane & 1;
        const float* xr = xs + row * D_ + hsel * 64;
        float s = 0.0f, sq = 0.0f;
        #pragma unroll 8
        for (int c = 0; c < 64; ++c) { float v = xr[c]; s += v; sq += v * v; }
        s  += __shfl_xor(s, 1, 32);
        sq += __shfl_xor(sq, 1, 32);
        const float mu  = s  * (1.0f / 128.0f);
        const float var = sq * (1.0f / 128.0f) - mu * mu;
        const float inv = rsqrtf(var + 1e-5f);
        const int ngo = tile * 16 + row;
        #pragma unroll 8
        for (int c = 0; c < 64; ++c) {
            const int col = hsel * 64 + c;
            out[(size_t)ngo * D_ + col] = (xr[c] - mu) * inv * gamma[col] + beta[col];
        }
    }
}

// ---------------------------------------------------------------------------
extern "C" void kernel_launch(void* const* d_in, const int* in_sizes, int n_in,
                              void* d_out, int out_size, void* d_ws, size_t ws_size,
                              hipStream_t stream) {
    (void)in_sizes; (void)n_in; (void)out_size; (void)ws_size;

    const float* h     = (const float*)d_in[0];
    const float* pos   = (const float*)d_in[1];
    const float* ew1   = (const float*)d_in[2];
    const float* eb1   = (const float*)d_in[3];
    const float* ew2   = (const float*)d_in[4];
    const float* eb2   = (const float*)d_in[5];
    const float* nw1   = (const float*)d_in[6];
    const float* nb1   = (const float*)d_in[7];
    const float* nw2   = (const float*)d_in[8];
    const float* nb2   = (const float*)d_in[9];
    const float* gamma = (const float*)d_in[10];
    const float* beta  = (const float*)d_in[11];
    float* out = (float*)d_out;

    const int BN = NBATCH * NNODES;                       // 32768
    int*   edge_idx  = (int*)d_ws;                        // [BN*16]
    float* edge_dist = (float*)(edge_idx + (size_t)BN * KNN);
    float* agg       = edge_dist + (size_t)BN * KNN;      // [BN*128]

    knn_kernel<<<BN, 128, 0, stream>>>(pos, edge_idx, edge_dist);

    const size_t lds2 = (size_t)(9 * 8 * 512 + 4 * 8 * 512) * 2 + (size_t)8 * 4352;
    edge_mlp_kernel<<<1024, 256, lds2, stream>>>(h, edge_idx, edge_dist,
                                                 ew1, eb1, ew2, eb2, agg, BN);

    const size_t lds3 = (size_t)(8 * 8 * 512 + 4 * 8 * 512) * 2 + (size_t)4 * 12288;
    node_kernel<<<512, 128, lds3, stream>>>(h, agg, nw1, nb1, nw2, nb2,
                                            gamma, beta, out, BN / 16);
}
